// Mamba3SISO_37177236914316
// MI455X (gfx1250) — compile-verified, matrix operand order checked by reading, not run on hardware
//
#include <hip/hip_runtime.h>
#include <hip/hip_bf16.h>
#include <math.h>
#include <stdint.h>

#define BATCH  4
#define SEQ    4096
#define DMODEL 2048
#define DSTATE 64
#define DINNER 1024
#define MROWS  (BATCH * SEQ)             // 16384
#define NTOT   (2 * DINNER + 2 * DSTATE) // 2176
#define CHL    64
#define NCH    (SEQ / CHL)               // 64

// LDS tile: 128 rows x 32 bf16, padded to 40 bf16 (80 B) per row
#define TROW   40
#define TBUFB  (128 * TROW * 2)          // bytes per buffer = 10240

typedef __attribute__((ext_vector_type(8)))  __bf16 v8bf;
typedef __attribute__((ext_vector_type(16))) __bf16 v16bf;
typedef __attribute__((ext_vector_type(8)))  float  v8f;

// ---- CDNA5 async global->LDS copy (16 bytes), tracked by ASYNCcnt ----
__device__ __forceinline__ void async_copy16(uint32_t lds_off, const __bf16* g) {
  asm volatile("global_load_async_to_lds_b128 %0, %1, off"
               :: "v"(lds_off), "v"(g)
               : "memory");
}
__device__ __forceinline__ void wait_async_all() {
  asm volatile("s_wait_asynccnt 0" ::: "memory");
}

// ---- fragment gather: 16 bf16 per lane as two 16B chunks ----
__device__ __forceinline__ v16bf load_frag(const __bf16* p) {
  v8bf a0 = *(const v8bf*)(p);
  v8bf a1 = *(const v8bf*)(p + 16);
  return __builtin_shufflevector(a0, a1, 0,1,2,3,4,5,6,7,8,9,10,11,12,13,14,15);
}

// =================== conversion kernels ===================
__global__ void k_cvt_f32_bf16(const float* __restrict__ src, __bf16* __restrict__ dst, int n) {
  int idx = blockIdx.x * blockDim.x + threadIdx.x;
  if (idx < n) dst[idx] = (__bf16)src[idx];
}

__global__ void k_build_wcat(const float* __restrict__ Wx, const float* __restrict__ Wdt,
                             const float* __restrict__ WB, const float* __restrict__ WC,
                             __bf16* __restrict__ wcat) {
  int idx = blockIdx.x * blockDim.x + threadIdx.x;   // NTOT*DMODEL
  if (idx >= NTOT * DMODEL) return;
  int r = idx / DMODEL, k = idx % DMODEL;
  float v;
  if (r < DINNER)            v = Wx [(size_t)r * DMODEL + k];
  else if (r < 2 * DINNER)   v = Wdt[(size_t)(r - DINNER) * DMODEL + k];
  else if (r < 2*DINNER+64)  v = WB [(size_t)(r - 2*DINNER) * DMODEL + k];
  else                       v = WC [(size_t)(r - 2*DINNER - 64) * DMODEL + k];
  wcat[idx] = (__bf16)v;
}

// =================== shared 128x128 GEMM tile body ===================
// 256 threads = 8 waves; wave (wm,wn) owns a 64(M) x 32(N) subtile -> acc[4][2].
// Double-buffered async global->LDS staging of 128x32 A/B slabs per K-step.
template <int LDK>
__device__ __forceinline__ void gemm_tile_128x128(
    const __bf16* __restrict__ A, const __bf16* __restrict__ Bw,
    int m0, int n0, v8f acc[4][2]) {
  __shared__ __align__(16) __bf16 As[2][128][TROW];
  __shared__ __align__(16) __bf16 Bs[2][128][TROW];

  const int tid  = threadIdx.x;
  const int lane = tid & 31;
  const int lo = lane & 15, hi = lane >> 4;
  const int w  = tid >> 5;
  const int wm = w & 1, wn = w >> 1;

  const uint32_t aBase = (uint32_t)(uintptr_t)&As[0][0][0];
  const uint32_t bBase = (uint32_t)(uintptr_t)&Bs[0][0][0];
  const int row0 = tid >> 2;        // 0..63 (thread covers row0 and row0+64)
  const int sub  = tid & 3;         // 16B chunk within 64B row

#pragma unroll
  for (int i = 0; i < 4; ++i)
#pragma unroll
    for (int j = 0; j < 2; ++j)
#pragma unroll
      for (int r = 0; r < 8; ++r) acc[i][j][r] = 0.0f;

  auto issue = [&](int buf, int k) {
    const uint32_t ab = aBase + buf * TBUFB;
    const uint32_t bb = bBase + buf * TBUFB;
    async_copy16(ab + (uint32_t)(row0      ) * (TROW*2) + sub * 16,
                 A + (size_t)(m0 + row0     ) * LDK + k + sub * 8);
    async_copy16(ab + (uint32_t)(row0 + 64 ) * (TROW*2) + sub * 16,
                 A + (size_t)(m0 + row0 + 64) * LDK + k + sub * 8);
    async_copy16(bb + (uint32_t)(row0      ) * (TROW*2) + sub * 16,
                 Bw + (size_t)(n0 + row0     ) * LDK + k + sub * 8);
    async_copy16(bb + (uint32_t)(row0 + 64 ) * (TROW*2) + sub * 16,
                 Bw + (size_t)(n0 + row0 + 64) * LDK + k + sub * 8);
  };

  issue(0, 0);
  const int KT = LDK / 32;
  for (int kt = 0; kt < KT; ++kt) {
    const int buf = kt & 1;
    wait_async_all();      // my slab for `buf` has landed in LDS
    __syncthreads();       // everyone's slab visible; prev readers of buf^1 done
    if (kt + 1 < KT) issue(buf ^ 1, (kt + 1) * 32);  // DMA next slab during compute

    v16bf af[4], bfg[2];
#pragma unroll
    for (int i = 0; i < 4; ++i)
      af[i] = load_frag(&As[buf][wm * 64 + i * 16 + lo][hi * 8]);
#pragma unroll
    for (int j = 0; j < 2; ++j)
      bfg[j] = load_frag(&Bs[buf][wn * 32 + j * 16 + lo][hi * 8]);
#pragma unroll
    for (int i = 0; i < 4; ++i)
#pragma unroll
      for (int j = 0; j < 2; ++j)
        acc[i][j] = __builtin_amdgcn_wmma_f32_16x16x32_bf16(
            false, af[i], false, bfg[j], (short)0, acc[i][j], false, false);
  }
}

// =================== fused projection GEMM ===================
// grid = (MROWS/128, NTOT/128) = (128, 17), 256 threads
__global__ __launch_bounds__(256)
void k_proj_gemm(const __bf16* __restrict__ xb, const __bf16* __restrict__ wcat,
                 const float* __restrict__ b_dt,
                 float* __restrict__ xssm, float* __restrict__ dtb,
                 float* __restrict__ bmo, float* __restrict__ cmo) {
  const int lane = threadIdx.x & 31;
  const int lo = lane & 15, hi = lane >> 4;
  const int w  = threadIdx.x >> 5;
  const int wm = w & 1, wn = w >> 1;
  const int m0 = blockIdx.x * 128;
  const int n0 = blockIdx.y * 128;

  v8f acc[4][2];
  gemm_tile_128x128<DMODEL>(xb, wcat, m0, n0, acc);

  // region boundaries (1024, 2048, 2112) are 64-aligned; branch per element
#pragma unroll
  for (int i = 0; i < 4; ++i)
#pragma unroll
    for (int j = 0; j < 2; ++j) {
      int col = n0 + wn * 32 + j * 16 + lo;
#pragma unroll
      for (int r = 0; r < 8; ++r) {
        int row = m0 + wm * 64 + i * 16 + hi * 8 + r;
        float v = acc[i][j][r];
        if (col < DINNER) {
          xssm[(size_t)row * DINNER + col] = v;
        } else if (col < 2 * DINNER) {
          int c2 = col - DINNER;
          float z = v + b_dt[c2];
          float sp = (z > 20.0f) ? z : log1pf(expf(z));
          sp = fminf(fmaxf(sp, 1e-4f), 0.5f);
          dtb[(size_t)row * DINNER + c2] = sp;
        } else if (col < 2 * DINNER + 64) {
          bmo[(size_t)row * 64 + (col - 2*DINNER)] = v;
        } else {
          cmo[(size_t)row * 64 + (col - 2*DINNER - 64)] = v;
        }
      }
    }
}

// =================== output GEMM: out = y @ Wout^T ===================
// grid = (MROWS/128, DINNER/128) = (128, 8), 256 threads
__global__ __launch_bounds__(256)
void k_out_gemm(const __bf16* __restrict__ yb, const __bf16* __restrict__ wb,
                float* __restrict__ out) {
  const int lane = threadIdx.x & 31;
  const int lo = lane & 15, hi = lane >> 4;
  const int w  = threadIdx.x >> 5;
  const int wm = w & 1, wn = w >> 1;
  const int m0 = blockIdx.x * 128;
  const int n0 = blockIdx.y * 128;

  v8f acc[4][2];
  gemm_tile_128x128<DINNER>(yb, wb, m0, n0, acc);

#pragma unroll
  for (int i = 0; i < 4; ++i)
#pragma unroll
    for (int j = 0; j < 2; ++j) {
      int col = n0 + wn * 32 + j * 16 + lo;
#pragma unroll
      for (int r = 0; r < 8; ++r) {
        int row = m0 + wm * 64 + i * 16 + hi * 8 + r;
        out[(size_t)row * DINNER + col] = acc[i][j][r];
      }
    }
}

// =================== row normalization of Bm / Cm ===================
// grid = (MROWS, 2), 64 threads
__global__ void k_rownorm(float* __restrict__ bm, float* __restrict__ cm) {
  float* ptr = (blockIdx.y == 0 ? bm : cm) + (size_t)blockIdx.x * 64;
  int t = threadIdx.x;
  float v = ptr[t];
  float s = v * v;
#pragma unroll
  for (int o = 16; o > 0; o >>= 1) s += __shfl_down(s, o, 32);
  __shared__ float sh[2];
  if ((t & 31) == 0) sh[t >> 5] = s;
  __syncthreads();
  float scale = 1.0f / fmaxf(sqrtf(sh[0] + sh[1]), 1.0f);
  ptr[t] = v * scale;
}

// =================== chunked scan, pass 1 ===================
__global__ void k_scan_chunk(const float* __restrict__ dtb, const float* __restrict__ xssm,
                             const float* __restrict__ bm, const float* __restrict__ A_log,
                             float* __restrict__ hbuf, float* __restrict__ cumA,
                             float* __restrict__ agg) {
  int tid = blockIdx.x * blockDim.x + threadIdx.x;   // BATCH*DINNER*NCH = 262144
  int i = tid % DINNER;
  int c = (tid / DINNER) % NCH;
  int b = tid / (DINNER * NCH);
  float an = -expf(A_log[i]);
  float h = 0.0f, cum = 1.0f;
  size_t base  = ((size_t)b * SEQ + (size_t)c * CHL) * DINNER + i;
  size_t baseB = ((size_t)b * SEQ + (size_t)c * CHL) * 64 + (i & 63);
  for (int j = 0; j < CHL; ++j) {
    float dtv = dtb[base + (size_t)j * DINNER];
    float xv  = xssm[base + (size_t)j * DINNER];
    float Bv  = bm[baseB + (size_t)j * 64];
    float a  = expf(dtv * an);
    float bb = dtv * Bv * xv;
    h = a * h + bb;
    cum *= a;
    hbuf[base + (size_t)j * DINNER] = h;
    cumA[base + (size_t)j * DINNER] = cum;
  }
  size_t aidx = ((size_t)b * DINNER + i) * NCH + c;
  agg[aidx * 2 + 0] = cum;
  agg[aidx * 2 + 1] = h;
}

// =================== scan pass 2: carry across chunks ===================
__global__ void k_scan_carry(const float* __restrict__ agg, float* __restrict__ pref) {
  int tid = blockIdx.x * blockDim.x + threadIdx.x;   // BATCH*DINNER = 4096
  if (tid >= BATCH * DINNER) return;
  float carry = 0.0f;
  size_t base = (size_t)tid * NCH;
  for (int c = 0; c < NCH; ++c) {
    pref[base + c] = carry;
    carry = agg[(base + c) * 2 + 0] * carry + agg[(base + c) * 2 + 1];
  }
}

// =================== scan pass 3: apply carry + y, emit bf16 ===================
__global__ void k_apply(const float* __restrict__ hbuf, const float* __restrict__ cumA,
                        const float* __restrict__ pref, const float* __restrict__ cmn,
                        const float* __restrict__ xssm, const float* __restrict__ Dp,
                        __bf16* __restrict__ ybuf) {
  size_t tid = (size_t)blockIdx.x * blockDim.x + threadIdx.x; // MROWS*DINNER
  int i = (int)(tid % DINNER);
  size_t m = tid / DINNER;
  int b = (int)(m / SEQ);
  int t = (int)(m % SEQ);
  int c = t / CHL;
  float carry = pref[((size_t)b * DINNER + i) * NCH + c];
  float h = hbuf[tid] + carry * cumA[tid];
  float y = cmn[m * 64 + (i & 63)] * h + Dp[i] * xssm[tid];
  ybuf[tid] = (__bf16)y;
}

// =================== launch ===================
extern "C" void kernel_launch(void* const* d_in, const int* in_sizes, int n_in,
                              void* d_out, int out_size, void* d_ws, size_t ws_size,
                              hipStream_t stream) {
  const float* x_norm = (const float*)d_in[0];
  const float* Wx     = (const float*)d_in[1];
  const float* Wdt    = (const float*)d_in[2];
  const float* b_dt   = (const float*)d_in[3];
  const float* WB     = (const float*)d_in[4];
  const float* WC     = (const float*)d_in[5];
  const float* A_log  = (const float*)d_in[6];
  const float* Dp     = (const float*)d_in[7];
  const float* Wout   = (const float*)d_in[8];
  float* out = (float*)d_out;

  char* ws = (char*)d_ws;
  size_t off = 0;
  __bf16* xb    = (__bf16*)(ws + off); off += (size_t)MROWS * DMODEL * 2;   // 64 MB
  __bf16* wcat  = (__bf16*)(ws + off); off += (size_t)NTOT * DMODEL * 2;    // 8.5 MB
  __bf16* woutb = (__bf16*)(ws + off); off += (size_t)DINNER * DINNER * 2;  // 2 MB
  float* xssm   = (float*)(ws + off);  off += (size_t)MROWS * DINNER * 4;   // 64 MB
  float* dtb    = (float*)(ws + off);  off += (size_t)MROWS * DINNER * 4;   // 64 MB
  float* bm     = (float*)(ws + off);  off += (size_t)MROWS * 64 * 4;       // 4 MB
  float* cm     = (float*)(ws + off);  off += (size_t)MROWS * 64 * 4;       // 4 MB
  float* hbuf   = (float*)(ws + off);  off += (size_t)MROWS * DINNER * 4;   // 64 MB
  float* cumA   = (float*)(ws + off);  off += (size_t)MROWS * DINNER * 4;   // 64 MB
  float* agg    = (float*)(ws + off);  off += (size_t)BATCH * DINNER * NCH * 2 * 4; // 2 MB
  float* pref   = (float*)(ws + off);  off += (size_t)BATCH * DINNER * NCH * 4;     // 1 MB
  __bf16* ybuf  = (__bf16*)(ws + off); off += (size_t)MROWS * DINNER * 2;   // 32 MB

  // 1) conversions
  {
    int n = MROWS * DMODEL;
    k_cvt_f32_bf16<<<(n + 255) / 256, 256, 0, stream>>>(x_norm, xb, n);
  }
  {
    int n = NTOT * DMODEL;
    k_build_wcat<<<(n + 255) / 256, 256, 0, stream>>>(Wx, Wdt, WB, WC, wcat);
  }
  {
    int n = DINNER * DINNER;
    k_cvt_f32_bf16<<<(n + 255) / 256, 256, 0, stream>>>(Wout, woutb, n);
  }

  // 2) fused projection GEMM (async-LDS staged, 8 waves/block)
  k_proj_gemm<<<dim3(MROWS / 128, NTOT / 128), 256, 0, stream>>>(
      xb, wcat, b_dt, xssm, dtb, bm, cm);

  // 3) normalize Bm, Cm rows
  k_rownorm<<<dim3(MROWS, 2), 64, 0, stream>>>(bm, cm);

  // 4-6) chunked associative scan
  k_scan_chunk<<<(BATCH * DINNER * NCH) / 256, 256, 0, stream>>>(
      dtb, xssm, bm, A_log, hbuf, cumA, agg);
  k_scan_carry<<<(BATCH * DINNER + 255) / 256, 256, 0, stream>>>(agg, pref);
  k_apply<<<((size_t)MROWS * DINNER) / 256, 256, 0, stream>>>(
      hbuf, cumA, pref, cm, xssm, Dp, ybuf);

  // 7) output GEMM
  k_out_gemm<<<dim3(MROWS / 128, DINNER / 128), 256, 0, stream>>>(ybuf, woutb, out);
}
